// LightningMLP_36584531427827
// MI455X (gfx1250) — compile-verified, matrix operand order checked by reading, not run on hardware
//
#include <hip/hip_runtime.h>

// ---------------------------------------------------------------------------
// Types for CDNA5 WMMA / TDM
// ---------------------------------------------------------------------------
typedef __attribute__((ext_vector_type(16))) __bf16 v16bf;
typedef __attribute__((ext_vector_type(8)))  __bf16 v8bf;
typedef __attribute__((ext_vector_type(8)))  float  v8f;
typedef __attribute__((ext_vector_type(4)))  unsigned int u32x4;
typedef __attribute__((ext_vector_type(8)))  int    i32x8;
typedef __attribute__((ext_vector_type(4)))  int    i32x4;

#define HIDDEN      256
#define SH_STRIDE   272          // bf16 elements per activation row (16 pad)
#define ROWS_TOTAL  131072       // 1*128*128*8
#define ROWS_PER_WG 64
#define THREADS     128          // 4 waves of 32

// Dynamic LDS layout: [0,34816) activations, then two 16KB weight buffers.
#define SMEM_H_BYTES   (ROWS_PER_WG * SH_STRIDE * 2)        // 34816
#define SMEM_W_BYTES   (256 * 32 * 2)                       // 16384
#define SMEM_TOTAL     (SMEM_H_BYTES + 2 * SMEM_W_BYTES)    // 67584

#ifndef __has_builtin
#define __has_builtin(x) 0
#endif
#define HAVE_TDM __has_builtin(__builtin_amdgcn_tensor_load_to_lds)

// ---------------------------------------------------------------------------
// Workspace layout (byte offsets into d_ws)
// ---------------------------------------------------------------------------
#define WS_WM     0              // bf16 [256][704]  (mask weights, K padded)
#define WS_W1     360448         // bf16 [256][64]
#define WS_W2     393216         // bf16 [256][256]
#define WS_W3     524288
#define WS_W4     655360
#define WS_W5     786432
#define WS_W6     917504
#define WS_VDPACK 1048576        // float4 [256]  {wv0*a5, wv1*a5, wv2*a5, beta5}
#define WS_ALPHA  1052672        // float  [5][256]  (a2,a3,a4,a6,a5)
#define WS_BETA   1057792        // float  [4][256]  (b2,b3,b4,b6)

// ---------------------------------------------------------------------------
// Prep: modulation vectors  alpha = z @ waT + ba,  beta = z @ wbT + bb
// ---------------------------------------------------------------------------
struct ModArgs {
    const float* z;
    const float* wa[5];
    const float* ba[5];
    const float* wb[5];
    const float* bb[5];
    const float* wv;          // fc_viewdir_w [256][3]
    float*       alpha[5];    // out
    float*       beta[4];     // out (fc2,fc3,fc4,fc6)
    float4*      vdpack;      // out
};

__global__ void prep_mod_kernel(ModArgs A) {
    const int t = threadIdx.x;          // output channel o (0..255)
    const float* z = A.z;
    for (int L = 0; L < 5; ++L) {
        float a = A.ba[L][t];
        float b = A.bb[L][t];
        const float* wa = A.wa[L] + t * 128;
        const float* wb = A.wb[L] + t * 128;
        for (int j = 0; j < 128; ++j) {
            float zj = z[j];
            a = fmaf(zj, wa[j], a);
            b = fmaf(zj, wb[j], b);
        }
        A.alpha[L][t] = a;
        if (L < 4) {
            A.beta[L][t] = b;
        } else {
            float4 v;
            v.x = A.wv[t * 3 + 0] * a;
            v.y = A.wv[t * 3 + 1] * a;
            v.z = A.wv[t * 3 + 2] * a;
            v.w = b;
            A.vdpack[t] = v;
        }
    }
}

// ---------------------------------------------------------------------------
// Prep: weight conversion / folding into bf16
// ---------------------------------------------------------------------------
__global__ void conv_pad_kernel(__bf16* dst, const float* src, int dstC, int srcC) {
    int i = blockIdx.x * 256 + threadIdx.x;
    int r = i / dstC;
    int c = i - r * dstC;
    dst[i] = (c < srcC) ? (__bf16)src[r * srcC + c] : (__bf16)0.0f;
}

__global__ void conv_scale_in_kernel(__bf16* dst, const float* src, const float* alpha) {
    int i = blockIdx.x * 256 + threadIdx.x;
    dst[i] = (__bf16)(src[i] * alpha[i & 255]);     // scale input channel (column)
}

__global__ void conv_scale_out_kernel(__bf16* dst, const float* src, const float* alpha) {
    int i = blockIdx.x * 256 + threadIdx.x;
    dst[i] = (__bf16)(src[i] * alpha[i >> 8]);      // scale output channel (row)
}

// ---------------------------------------------------------------------------
// Main fused MLP kernel
// ---------------------------------------------------------------------------
struct NetArgs {
    const float*  x;         // [rows][64]
    const float*  raydir;    // [16384][3]
    const float*  m;         // [rows][680]
    const float*  fc1_b;     // [256]
    const float*  sigma_w;   // [256]
    const float*  sigma_b;   // [1]
    const float*  out_c_w;   // [3][256]
    const float*  out_c_b;   // [3]
    const __bf16* W1;        // [256][64]
    const __bf16* Wm;        // [256][704]
    const __bf16* W2;        // [256][256] each
    const __bf16* W3;
    const __bf16* W4;
    const __bf16* W5;
    const __bf16* W6;
    const float*  beta2;
    const float*  beta3;
    const float*  beta4;
    const float*  beta6;
    const float4* vdpack;
    float*        out;       // sigma[131072] then c[131072*3]
};

__device__ __forceinline__ float lrelu(float v) { return v > 0.0f ? v : 0.2f * v; }

__device__ __forceinline__ v16bf pack16(v8bf lo, v8bf hi) {
    v16bf a;
#pragma unroll
    for (int i = 0; i < 8; ++i) { a[i] = lo[i]; a[i + 8] = hi[i]; }
    return a;
}

// A fragment (16x32 bf16) from LDS activation rows.
__device__ __forceinline__ v16bf loadA_lds(const __bf16* sh_h, int rowLocal, int k0, int hsel) {
    const __bf16* p = sh_h + rowLocal * SH_STRIDE + k0 + hsel;
    v8bf lo = *(const v8bf*)p;
    v8bf hi = *(const v8bf*)(p + 16);
    return pack16(lo, hi);
}

// B fragment (32x16 bf16) for column o from the [256][32] LDS weight chunk.
__device__ __forceinline__ v16bf loadB_lds(const __bf16* sh_w, int o, int hsel) {
    const __bf16* p = sh_w + o * 32 + hsel;
    v8bf lo = *(const v8bf*)p;
    v8bf hi = *(const v8bf*)(p + 16);
    return pack16(lo, hi);
}

// ---------------------------------------------------------------------------
// Weight-chunk staging: TDM (tensor_load_to_lds) with double buffering,
// falling back to a cooperative VMEM+DS loader if the builtin is absent.
// One 2-D tile: 256 rows x 32 bf16 columns from a row-major [256][stride].
// ---------------------------------------------------------------------------
#if HAVE_TDM
__device__ __forceinline__ void tdm_load_chunk(const __bf16* W, int wstrideElems, int k0,
                                               __bf16* ldsDst) {
    unsigned long long ga = (unsigned long long)(const void*)(W + k0); // tile start (bytes)
    unsigned int lds = (unsigned int)(unsigned long long)(void*)ldsDst; // LDS aperture: low 32b

    u32x4 g0;
    g0[0] = 1u;                                   // count=1, user descriptor
    g0[1] = lds;                                  // lds_addr
    g0[2] = (unsigned int)ga;                     // global_addr[31:0]
    g0[3] = (unsigned int)(ga >> 32) | (2u << 30);// global_addr[56:32] | type=2 (image)

    i32x8 g1;
    g1[0] = 1 << 16;                              // workgroup_mask=0, data_size=1 (2B)
    g1[1] = (wstrideElems & 0xffff) << 16;        // tensor_dim0[15:0]  (bits 63:48)
    g1[2] = ((wstrideElems >> 16) & 0xffff)       // tensor_dim0[31:16]
          | (256 << 16);                          // tensor_dim1[15:0] = 256 rows
    g1[3] = (32 << 16);                           // tensor_dim1[31:16]=0 | tile_dim0=32
    g1[4] = 256;                                  // tile_dim1=256 | tile_dim2=0
    g1[5] = wstrideElems;                         // tensor_dim0_stride[31:0]
    g1[6] = 0;                                    // stride hi | dim1_stride lo
    g1[7] = 0;

    i32x4 gz;  gz[0] = gz[1] = gz[2] = gz[3] = 0; // groups 2/3 unused (2-D tile)
#if defined(__clang_major__) && __clang_major__ >= 23
    i32x8 gz8;
#pragma unroll
    for (int i = 0; i < 8; ++i) gz8[i] = 0;
    __builtin_amdgcn_tensor_load_to_lds(g0, g1, gz, gz, gz8, 0);
#else
    __builtin_amdgcn_tensor_load_to_lds(g0, g1, gz, gz, 0);
#endif
}
#else
// Fallback: cooperative 16KB copy with all 128 threads.
__device__ __forceinline__ void coop_load_chunk(__bf16* sh_w, const __bf16* W, int wstride,
                                                int k0, int tid) {
#pragma unroll
    for (int i = 0; i < 8; ++i) {
        int idx = tid + i * THREADS;          // 0..1023
        int row = idx >> 2;
        int seg = idx & 3;
        *(uint4*)(sh_w + row * 32 + seg * 8) =
            *(const uint4*)(W + row * wstride + k0 + seg * 8);
    }
}
#endif

__device__ __forceinline__ void load8f(float* f, const float* rowp, int off, int kmax) {
    if (off + 8 <= kmax) {
        float4 a = *(const float4*)(rowp + off);
        float4 b = *(const float4*)(rowp + off + 4);
        f[0] = a.x; f[1] = a.y; f[2] = a.z; f[3] = a.w;
        f[4] = b.x; f[5] = b.y; f[6] = b.z; f[7] = b.w;
    } else {
#pragma unroll
        for (int j = 0; j < 8; ++j) {
            int k = off + j;
            f[j] = (k < kmax) ? rowp[k] : 0.0f;
        }
    }
}

__device__ __forceinline__ v16bf loadA_glb(const float* Am, int C, int row, int k0,
                                           int kmax, int hsel) {
    const float* rowp = Am + (size_t)row * C;
    float f[16];
    load8f(f,     rowp, k0 + hsel,      kmax);
    load8f(f + 8, rowp, k0 + hsel + 16, kmax);
    v16bf a;
#pragma unroll
    for (int i = 0; i < 16; ++i) a[i] = (__bf16)f[i];
    return a;
}

// 16 n-tiles of WMMA with the B fragments software-pipelined 2 deep so the
// scheduler can keep >=2 ds_load_b128 pairs in flight per wmma (avoids the
// single-register s_wait_dscnt 0 serialization).
__device__ __forceinline__ void wmma16(v8f acc[16], v16bf a, const __bf16* sh_w,
                                       int l15, int hsel) {
    v16bf b0 = loadB_lds(sh_w, 0 * 16 + l15, hsel);
    v16bf b1 = loadB_lds(sh_w, 1 * 16 + l15, hsel);
#pragma unroll
    for (int nt = 0; nt < 14; ++nt) {
        v16bf bn = loadB_lds(sh_w, (nt + 2) * 16 + l15, hsel);
        acc[nt] = __builtin_amdgcn_wmma_f32_16x16x32_bf16(
            false, a, false, b0, (short)0, acc[nt], false, false);
        b0 = b1;
        b1 = bn;
    }
    acc[14] = __builtin_amdgcn_wmma_f32_16x16x32_bf16(
        false, a, false, b0, (short)0, acc[14], false, false);
    acc[15] = __builtin_amdgcn_wmma_f32_16x16x32_bf16(
        false, a, false, b1, (short)0, acc[15], false, false);
}

// Stage chunk kc+1 while computing chunk kc (TDM path); buffers ping-pong.
// nchunks is even for every layer, so a new layer's prologue into buf0 never
// races the previous layer's final chunk (which reads buf1).
__device__ __forceinline__ void stage_and_wait(const __bf16* W, int wstride, int kc,
                                               int nchunks, __bf16* buf0, __bf16* buf1,
                                               bool issuer, int tid) {
#if HAVE_TDM
    if (kc + 1 < nchunks) {
        if (issuer) {
            tdm_load_chunk(W, wstride, (kc + 1) * 32, ((kc + 1) & 1) ? buf1 : buf0);
            __builtin_amdgcn_s_wait_tensorcnt(1);   // chunk kc complete
        }
    } else {
        if (issuer) __builtin_amdgcn_s_wait_tensorcnt(0);
    }
#else
    coop_load_chunk((kc & 1) ? buf1 : buf0, W, wstride, kc * 32, tid);
#endif
}

__device__ __forceinline__ void gemm_lds(v8f acc[16], const __bf16* sh_h,
                                         __bf16* buf0, __bf16* buf1,
                                         const __bf16* W, int rowLocal, int l15, int hsel,
                                         bool issuer, int tid) {
#if HAVE_TDM
    if (issuer) tdm_load_chunk(W, 256, 0, buf0);
#endif
    for (int kc = 0; kc < 8; ++kc) {
        __syncthreads();                                   // buf[(kc+1)&1] free
        stage_and_wait(W, 256, kc, 8, buf0, buf1, issuer, tid);
        __syncthreads();                                   // chunk kc visible
        v16bf a = loadA_lds(sh_h, rowLocal, kc * 32, hsel);
        wmma16(acc, a, (kc & 1) ? buf1 : buf0, l15, hsel);
    }
}

__device__ __forceinline__ void gemm_glb(v8f acc[16], const float* Am, int C, int kmax,
                                         __bf16* buf0, __bf16* buf1,
                                         const __bf16* W, int wstride, int nchunks,
                                         int aRow, int l15, int hsel, bool issuer, int tid) {
#if HAVE_TDM
    if (issuer) tdm_load_chunk(W, wstride, 0, buf0);
#endif
    for (int kc = 0; kc < nchunks; ++kc) {
        __syncthreads();
        stage_and_wait(W, wstride, kc, nchunks, buf0, buf1, issuer, tid);
        __syncthreads();
        v16bf a = loadA_glb(Am, C, aRow, kc * 32, kmax, hsel);
        wmma16(acc, a, (kc & 1) ? buf1 : buf0, l15, hsel);
    }
}

__device__ __forceinline__ void zero_acc(v8f acc[16]) {
#pragma unroll
    for (int nt = 0; nt < 16; ++nt)
#pragma unroll
        for (int r = 0; r < 8; ++r) acc[nt][r] = 0.0f;
}

__device__ __forceinline__ void epilogue_act_store(v8f acc[16], const float* bias,
                                                   __bf16* sh_h, int wRow0, int l15) {
#pragma unroll
    for (int nt = 0; nt < 16; ++nt) {
        int o = nt * 16 + l15;
        float b = bias[o];
#pragma unroll
        for (int r = 0; r < 8; ++r) {
            float v = lrelu(acc[nt][r] + b);
            sh_h[(wRow0 + r) * SH_STRIDE + o] = (__bf16)v;
        }
    }
}

// Sum over the 16 lanes of each wave32 half (xor masks 1,2,4,8 never cross bit4).
__device__ __forceinline__ void red16_8(float v[8]) {
#pragma unroll
    for (int r = 0; r < 8; ++r) {
        v[r] += __shfl_xor(v[r], 1, 32);
        v[r] += __shfl_xor(v[r], 2, 32);
        v[r] += __shfl_xor(v[r], 4, 32);
        v[r] += __shfl_xor(v[r], 8, 32);
    }
}

__global__ __launch_bounds__(THREADS) void mlp_main_kernel(NetArgs A) {
    extern __shared__ __align__(16) char smem[];
    __bf16* sh_h  = (__bf16*)smem;                          // 34816 B
    __bf16* buf0  = (__bf16*)(smem + SMEM_H_BYTES);         // 16384 B
    __bf16* buf1  = buf0 + 256 * 32;                        // 16384 B

    const int tid  = threadIdx.x;
    const int wave = tid >> 5;
    const int lane = tid & 31;
    const int l15  = lane & 15;
    const int hsel = (lane & 16) ? 8 : 0;   // K-half select for A/B fragments
    const int hrow = (lane & 16) ? 8 : 0;   // row-half select for C fragments
    const int rowBase  = blockIdx.x * ROWS_PER_WG + wave * 16;
    const int aRow     = rowBase + l15;        // global A row for this lane
    const int rowLocal = wave * 16 + l15;      // LDS A row
    const int wRow0    = wave * 16 + hrow;     // LDS C writeback row start
    const bool issuer  = (wave == 0);          // TDM issuing wave

    v8f acc[16];

    // ---- Layer 1: x @ W1^T + m @ Wm^T + b1, LeakyReLU --------------------
    zero_acc(acc);
    gemm_glb(acc, A.x, 64,  64,  buf0, buf1, A.W1, 64,  2,  aRow, l15, hsel, issuer, tid);
    gemm_glb(acc, A.m, 680, 680, buf0, buf1, A.Wm, 704, 22, aRow, l15, hsel, issuer, tid);
    epilogue_act_store(acc, A.fc1_b, sh_h, wRow0, l15);

    // ---- fc2 / fc3 (modulation folded into W, bias=beta) -----------------
    zero_acc(acc);
    gemm_lds(acc, sh_h, buf0, buf1, A.W2, rowLocal, l15, hsel, issuer, tid);
    epilogue_act_store(acc, A.beta2, sh_h, wRow0, l15);

    zero_acc(acc);
    gemm_lds(acc, sh_h, buf0, buf1, A.W3, rowLocal, l15, hsel, issuer, tid);
    epilogue_act_store(acc, A.beta3, sh_h, wRow0, l15);

    // ---- fc4 (+ sigma head) ---------------------------------------------
    zero_acc(acc);
    gemm_lds(acc, sh_h, buf0, buf1, A.W4, rowLocal, l15, hsel, issuer, tid);
    {
        float sp[8];
#pragma unroll
        for (int r = 0; r < 8; ++r) sp[r] = 0.0f;
#pragma unroll
        for (int nt = 0; nt < 16; ++nt) {
            int o = nt * 16 + l15;
            float b  = A.beta4[o];
            float sw = A.sigma_w[o];
#pragma unroll
            for (int r = 0; r < 8; ++r) {
                float v = lrelu(acc[nt][r] + b);
                sh_h[(wRow0 + r) * SH_STRIDE + o] = (__bf16)v;
                sp[r] = fmaf(v, sw, sp[r]);
            }
        }
        red16_8(sp);
        if (l15 == 0) {
            float sb = A.sigma_b[0];
#pragma unroll
            for (int r = 0; r < 8; ++r)
                A.out[rowBase + hrow + r] = sp[r] + sb;
        }
    }

    // ---- fc5 (alpha5 folded into W5) + viewdir + beta5, LeakyReLU --------
    zero_acc(acc);
    gemm_lds(acc, sh_h, buf0, buf1, A.W5, rowLocal, l15, hsel, issuer, tid);
    {
        float r0[8], r1[8], r2[8];
#pragma unroll
        for (int r = 0; r < 8; ++r) {
            int pix = (rowBase + hrow + r) >> 3;   // M=8 samples per pixel
            r0[r] = A.raydir[pix * 3 + 0];
            r1[r] = A.raydir[pix * 3 + 1];
            r2[r] = A.raydir[pix * 3 + 2];
        }
#pragma unroll
        for (int nt = 0; nt < 16; ++nt) {
            int o = nt * 16 + l15;
            float4 vw = A.vdpack[o];
#pragma unroll
            for (int r = 0; r < 8; ++r) {
                float v = acc[nt][r] + vw.w;
                v = fmaf(r0[r], vw.x, v);
                v = fmaf(r1[r], vw.y, v);
                v = fmaf(r2[r], vw.z, v);
                v = lrelu(v);
                sh_h[(wRow0 + r) * SH_STRIDE + o] = (__bf16)v;
            }
        }
    }

    // ---- fc6 (+ color head) ---------------------------------------------
    zero_acc(acc);
    gemm_lds(acc, sh_h, buf0, buf1, A.W6, rowLocal, l15, hsel, issuer, tid);
    {
        float c0[8], c1[8], c2[8];
#pragma unroll
        for (int r = 0; r < 8; ++r) { c0[r] = 0.0f; c1[r] = 0.0f; c2[r] = 0.0f; }
#pragma unroll
        for (int nt = 0; nt < 16; ++nt) {
            int o = nt * 16 + l15;
            float b  = A.beta6[o];
            float w0 = A.out_c_w[o];
            float w1 = A.out_c_w[256 + o];
            float w2 = A.out_c_w[512 + o];
#pragma unroll
            for (int r = 0; r < 8; ++r) {
                float v = lrelu(acc[nt][r] + b);
                c0[r] = fmaf(v, w0, c0[r]);
                c1[r] = fmaf(v, w1, c1[r]);
                c2[r] = fmaf(v, w2, c2[r]);
            }
        }
        red16_8(c0); red16_8(c1); red16_8(c2);
        if (l15 == 0) {
            float b0 = A.out_c_b[0], b1 = A.out_c_b[1], b2 = A.out_c_b[2];
#pragma unroll
            for (int r = 0; r < 8; ++r) {
                int gr = rowBase + hrow + r;
                float* op = A.out + ROWS_TOTAL + (size_t)gr * 3;
                op[0] = c0[r] + b0;
                op[1] = c1[r] + b1;
                op[2] = c2[r] + b2;
            }
        }
    }
}

// ---------------------------------------------------------------------------
// Host entry
// ---------------------------------------------------------------------------
extern "C" void kernel_launch(void* const* d_in, const int* in_sizes, int n_in,
                              void* d_out, int out_size, void* d_ws, size_t ws_size,
                              hipStream_t stream) {
    (void)in_sizes; (void)n_in; (void)out_size; (void)ws_size;

    // setup_inputs() dict order (note fc6 comes before sigma/fc5/mod5)
    const float* x            = (const float*)d_in[0];
    const float* raydir       = (const float*)d_in[1];
    const float* z            = (const float*)d_in[2];
    const float* m            = (const float*)d_in[3];
    const float* fc_m_a_w     = (const float*)d_in[4];
    const float* fc_viewdir_w = (const float*)d_in[5];
    const float* fc1_w        = (const float*)d_in[6];
    const float* fc1_b        = (const float*)d_in[7];
    const float* fc2_w  = (const float*)d_in[8];
    const float* fc2_wa = (const float*)d_in[9];
    const float* fc2_ba = (const float*)d_in[10];
    const float* fc2_wb = (const float*)d_in[11];
    const float* fc2_bb = (const float*)d_in[12];
    const float* fc3_w  = (const float*)d_in[13];
    const float* fc3_wa = (const float*)d_in[14];
    const float* fc3_ba = (const float*)d_in[15];
    const float* fc3_wb = (const float*)d_in[16];
    const float* fc3_bb = (const float*)d_in[17];
    const float* fc4_w  = (const float*)d_in[18];
    const float* fc4_wa = (const float*)d_in[19];
    const float* fc4_ba = (const float*)d_in[20];
    const float* fc4_wb = (const float*)d_in[21];
    const float* fc4_bb = (const float*)d_in[22];
    const float* fc6_w  = (const float*)d_in[23];
    const float* fc6_wa = (const float*)d_in[24];
    const float* fc6_ba = (const float*)d_in[25];
    const float* fc6_wb = (const float*)d_in[26];
    const float* fc6_bb = (const float*)d_in[27];
    const float* fc_sigma_w = (const float*)d_in[28];
    const float* fc_sigma_b = (const float*)d_in[29];
    const float* fc5_w      = (const float*)d_in[30];
    const float* mod5_wa    = (const float*)d_in[31];
    const float* mod5_ba    = (const float*)d_in[32];
    const float* mod5_wb    = (const float*)d_in[33];
    const float* mod5_bb    = (const float*)d_in[34];
    const float* fc_out_c_w = (const float*)d_in[35];
    const float* fc_out_c_b = (const float*)d_in[36];

    char* ws = (char*)d_ws;
    __bf16* Wm = (__bf16*)(ws + WS_WM);
    __bf16* W1 = (__bf16*)(ws + WS_W1);
    __bf16* W2 = (__bf16*)(ws + WS_W2);
    __bf16* W3 = (__bf16*)(ws + WS_W3);
    __bf16* W4 = (__bf16*)(ws + WS_W4);
    __bf16* W5 = (__bf16*)(ws + WS_W5);
    __bf16* W6 = (__bf16*)(ws + WS_W6);
    float4* vdpack = (float4*)(ws + WS_VDPACK);
    float*  alpha  = (float*)(ws + WS_ALPHA);   // a2,a3,a4,a6,a5
    float*  beta   = (float*)(ws + WS_BETA);    // b2,b3,b4,b6

    // --- modulation vectors ---
    ModArgs M{};
    M.z = z; M.wv = fc_viewdir_w;
    M.wa[0] = fc2_wa; M.ba[0] = fc2_ba; M.wb[0] = fc2_wb; M.bb[0] = fc2_bb;
    M.wa[1] = fc3_wa; M.ba[1] = fc3_ba; M.wb[1] = fc3_wb; M.bb[1] = fc3_bb;
    M.wa[2] = fc4_wa; M.ba[2] = fc4_ba; M.wb[2] = fc4_wb; M.bb[2] = fc4_bb;
    M.wa[3] = fc6_wa; M.ba[3] = fc6_ba; M.wb[3] = fc6_wb; M.bb[3] = fc6_bb;
    M.wa[4] = mod5_wa; M.ba[4] = mod5_ba; M.wb[4] = mod5_wb; M.bb[4] = mod5_bb;
    for (int L = 0; L < 5; ++L) M.alpha[L] = alpha + L * 256;
    for (int L = 0; L < 4; ++L) M.beta[L]  = beta  + L * 256;
    M.vdpack = vdpack;
    prep_mod_kernel<<<1, 256, 0, stream>>>(M);

    // --- weight fold + bf16 conversion ---
    conv_pad_kernel<<<704, 256, 0, stream>>>(Wm, fc_m_a_w, 704, 680);
    conv_pad_kernel<<<64,  256, 0, stream>>>(W1, fc1_w,    64,  64);
    conv_scale_in_kernel <<<256, 256, 0, stream>>>(W2, fc2_w, alpha + 0);
    conv_scale_in_kernel <<<256, 256, 0, stream>>>(W3, fc3_w, alpha + 256);
    conv_scale_in_kernel <<<256, 256, 0, stream>>>(W4, fc4_w, alpha + 512);
    conv_scale_in_kernel <<<256, 256, 0, stream>>>(W6, fc6_w, alpha + 768);
    conv_scale_out_kernel<<<256, 256, 0, stream>>>(W5, fc5_w, alpha + 1024);

    // --- fused MLP ---
    NetArgs N{};
    N.x = x; N.raydir = raydir; N.m = m;
    N.fc1_b = fc1_b; N.sigma_w = fc_sigma_w; N.sigma_b = fc_sigma_b;
    N.out_c_w = fc_out_c_w; N.out_c_b = fc_out_c_b;
    N.W1 = W1; N.Wm = Wm; N.W2 = W2; N.W3 = W3; N.W4 = W4; N.W5 = W5; N.W6 = W6;
    N.beta2 = beta; N.beta3 = beta + 256; N.beta4 = beta + 512; N.beta6 = beta + 768;
    N.vdpack = vdpack;
    N.out = (float*)d_out;
    mlp_main_kernel<<<ROWS_TOTAL / ROWS_PER_WG, THREADS, SMEM_TOTAL, stream>>>(N);
}
